// XFormersAttention_83854941487521
// MI455X (gfx1250) — compile-verified, matrix operand order checked by reading, not run on hardware
//
#include <hip/hip_runtime.h>
#include <hip/hip_bf16.h>

typedef __attribute__((ext_vector_type(16))) __bf16 v16bf;
typedef __attribute__((ext_vector_type(8)))  __bf16 v8bf;
typedef __attribute__((ext_vector_type(8)))  float  v8f;
typedef __attribute__((ext_vector_type(4)))  float  v4f;

constexpr int B_ = 4;
constexpr int S_ = 2048;
constexpr int E_ = 256;
constexpr int H_ = 8;
constexpr int D_ = 32;
#define SCALE_ 0.17677669529663687f   // 32^-0.5

// Hardware bf16 convert (RNE) — lowers to v_cvt_pk_bf16_f32 on gfx1250.
__device__ __forceinline__ __bf16 f2bf(float f) { return (__bf16)f; }

__device__ __forceinline__ v16bf cat8(v8bf lo, v8bf hi) {
    return __builtin_shufflevector(lo, hi, 0,1,2,3,4,5,6,7,8,9,10,11,12,13,14,15);
}

// Load a bf16 A/B fragment: 8 contiguous halves at p, 8 more at p+16 elements.
__device__ __forceinline__ v16bf load_frag_bf16(const __bf16* p) {
    v8bf lo = *(const v8bf*)(p);
    v8bf hi = *(const v8bf*)(p + 16);
    return cat8(lo, hi);
}

// Build a bf16 fragment from fp32 memory (8 floats at p, 8 at p+16).
__device__ __forceinline__ v16bf load_frag_f32(const float* p) {
    v16bf r;
#pragma unroll
    for (int j = 0; j < 8; ++j) r[j] = f2bf(p[j]);
#pragma unroll
    for (int j = 0; j < 8; ++j) r[8 + j] = f2bf(p[16 + j]);
    return r;
}

// CDNA5 async global->LDS staging (ASYNCcnt-tracked), per ISA §15.18.3.
__device__ __forceinline__ void async_load_b128(unsigned lds_off, const void* gaddr) {
    asm volatile("global_load_async_to_lds_b128 %0, %1, off"
                 :: "v"(lds_off), "v"(gaddr) : "memory");
}
__device__ __forceinline__ void wait_asynccnt0() {
    asm volatile("s_wait_asynccnt 0x0" ::: "memory");
}
__device__ __forceinline__ void wait_dscnt0() {
    asm volatile("s_wait_dscnt 0x0" ::: "memory");
}

// ---------------- kernel 0: per-sample valid lengths (prefix-mask sum) ------
__global__ void lengths_kernel(const int* __restrict__ mask, int* __restrict__ lengths) {
    const int b = blockIdx.x;
    const int t = threadIdx.x;
    int sum = 0;
    for (int i = t; i < S_; i += 256) sum += mask[b * S_ + i];
#pragma unroll
    for (int off = 16; off > 0; off >>= 1) sum += __shfl_down(sum, off, 32);
    __shared__ int part[8];
    if ((t & 31) == 0) part[t >> 5] = sum;
    __syncthreads();
    if (t == 0) {
        int tot = 0;
#pragma unroll
        for (int i = 0; i < 8; ++i) tot += part[i];
        lengths[b] = tot;
    }
}

// ---------------- kernel 1: QKV projections via WMMA ------------------------
// y = x @ W.T ; C tiles staged through LDS so global stores are coalesced b128.
__global__ __launch_bounds__(256) void qkv_gemm_kernel(
        const float* __restrict__ x,
        const float* __restrict__ Wq, const float* __restrict__ Wk, const float* __restrict__ Wv,
        __bf16* __restrict__ Qb, __bf16* __restrict__ Kb, __bf16* __restrict__ Vt)
{
    const int rt    = blockIdx.x;        // 512 row tiles of 16 over B*S
    const int which = blockIdx.z;        // 0=Q, 1=K, 2=V
    const int lane  = threadIdx.x;       // 0..31
    const int wy    = threadIdx.y;       // 0..7 -> col tiles wy and wy+8
    const float* W  = (which == 0) ? Wq : (which == 1) ? Wk : Wv;

    const int m     = lane & 15;
    const int koff  = (lane >> 4) * 8;
    const int nloc  = lane & 15;
    const long rowIdx = (long)rt * 16 + m;
    const int ct[2] = { wy, wy + 8 };

    v8f acc[2] = { {}, {} };
    for (int kk = 0; kk < E_; kk += 32) {
        v16bf a  = load_frag_f32(x + rowIdx * E_ + kk + koff);
#pragma unroll
        for (int c = 0; c < 2; ++c) {
            v16bf bfr = load_frag_f32(W + (long)(ct[c] * 16 + nloc) * E_ + kk + koff);
            acc[c] = __builtin_amdgcn_wmma_f32_16x16x32_bf16(false, a, false, bfr,
                                                             (short)0, acc[c], false, false);
        }
    }

    // Stage C tiles in LDS (bf16); V is staged transposed so the [d][s] store
    // below is contiguous.  8 waves x 2 tiles x 16x16 bf16 = 8 KB.
    __shared__ alignas(16) __bf16 Ctile[8][2][16][16];
    const int mbase = (lane >> 4) * 8;
    if (which == 2) {
        // [d][s] layout: this lane's 8 values are contiguous -> one b128 store.
#pragma unroll
        for (int c = 0; c < 2; ++c) {
            v8bf pk;
#pragma unroll
            for (int r = 0; r < 8; ++r) pk[r] = f2bf(acc[c][r]);
            *(v8bf*)&Ctile[wy][c][nloc][mbase] = pk;
        }
    } else {
        // [s][d] layout: column writes (b16), straight-line.
#pragma unroll
        for (int c = 0; c < 2; ++c)
#pragma unroll
            for (int r = 0; r < 8; ++r)
                Ctile[wy][c][mbase + r][nloc] = f2bf(acc[c][r]);
    }
    __syncthreads();

    const int row0 = rt * 16;
    const int b    = row0 >> 11;           // whole tile shares one sample
    const int s0   = row0 & (S_ - 1);
    const int tm   = lane >> 1;            // tile row handled by this lane
    const int half = lane & 1;             // which 16B chunk of the 32B row
    if (which == 2) {
#pragma unroll
        for (int c = 0; c < 2; ++c) {
            const int h  = ct[c] >> 1;
            const int d0 = (ct[c] & 1) * 16;
            const v8bf v = *(const v8bf*)&Ctile[wy][c][tm][half * 8];
            __bf16* dst = Vt + ((size_t)(b * H_ + h) * D_ + d0 + tm) * S_ + s0 + half * 8;
            *(v8bf*)dst = v;
        }
    } else {
        __bf16* base = (which == 0) ? Qb : Kb;
#pragma unroll
        for (int c = 0; c < 2; ++c) {
            const int h  = ct[c] >> 1;
            const int d0 = (ct[c] & 1) * 16;
            const v8bf v = *(const v8bf*)&Ctile[wy][c][tm][half * 8];
            __bf16* dst = base + ((size_t)(b * H_ + h) * S_ + s0 + tm) * D_ + d0 + half * 8;
            *(v8bf*)dst = v;
        }
    }
}

// ---------------- kernel 2: flash attention, one wave per (b,h,16-row tile) -
__global__ __launch_bounds__(32) void attn_kernel(
        const __bf16* __restrict__ Qb, const __bf16* __restrict__ Kb,
        const __bf16* __restrict__ Vt, const int* __restrict__ lengths,
        __bf16* __restrict__ attnb)
{
    const int qt = blockIdx.x, h = blockIdx.y, b = blockIdx.z;
    const int lane = threadIdx.x;
    const int q0 = qt * 16;
    const int bh = b * H_ + h;
    const int m     = lane & 15;
    const int koff  = (lane >> 4) * 8;
    const int nloc  = lane & 15;
    const int mbase = (lane >> 4) * 8;

    __shared__ alignas(16) __bf16 Klds[32][32];   // 32 keys x 32 d, async-staged
    __shared__ alignas(16) __bf16 Plds[16][32];   // P tile relayout + output staging

    // Q fragment (A layout), K-dim = full head dim D=32
    const v16bf aQ = load_frag_bf16(Qb + ((size_t)bh * S_ + q0 + m) * D_ + koff);

    float mi[8], li[8];
    v8f o0 = {}, o1 = {};
#pragma unroll
    for (int r = 0; r < 8; ++r) { mi[r] = -__builtin_inff(); li[r] = 0.0f; }

    const int L   = lengths[b];
    const int nkb = (L + 31) >> 5;          // skipped blocks contribute exactly 0

    for (int kb = 0; kb < nkb; ++kb) {
        const int k0 = kb * 32;
        if (kb + 1 < nkb)  // pull next V block toward the caches
            __builtin_prefetch(Vt + ((size_t)bh * D_ + nloc) * S_ + k0 + 32, 0, 3);

        // Async-stage the 32x32 K tile into LDS: 128 chunks of 16B, 4 per lane.
        wait_dscnt0();                       // prior ds reads of Klds are done
#pragma unroll
        for (int c = 0; c < 4; ++c) {
            const int id   = c * 32 + lane;
            const int key  = id >> 2;
            const int part = id & 3;
            const __bf16* g = Kb + ((size_t)bh * S_ + k0 + key) * D_ + part * 8;
            async_load_b128((unsigned)(size_t)&Klds[key][part * 8], g);
        }
        wait_asynccnt0();
        __syncthreads();

        // scores: Q[16x32] x K^T[32 x 32keys] -> two 16x16 C tiles (B-frags from LDS)
        v8f sc[2];
#pragma unroll
        for (int t = 0; t < 2; ++t) {
            v16bf bK = load_frag_bf16(&Klds[t * 16 + nloc][koff]);
            v8f z = {};
            sc[t] = __builtin_amdgcn_wmma_f32_16x16x32_bf16(false, aQ, false, bK,
                                                            (short)0, z, false, false);
        }

        float alpha[8];
#pragma unroll
        for (int r = 0; r < 8; ++r) {
            float s0 = sc[0][r] * SCALE_; if (k0 + nloc      >= L) s0 = -1e30f;
            float s1 = sc[1][r] * SCALE_; if (k0 + 16 + nloc >= L) s1 = -1e30f;
            float mx = fmaxf(s0, s1);
            mx = fmaxf(mx, __shfl_xor(mx, 1, 32));
            mx = fmaxf(mx, __shfl_xor(mx, 2, 32));
            mx = fmaxf(mx, __shfl_xor(mx, 4, 32));
            mx = fmaxf(mx, __shfl_xor(mx, 8, 32));   // row max within 16-lane half
            const float mnew = fmaxf(mi[r], mx);
            alpha[r] = __expf(mi[r] - mnew);
            const float e0 = __expf(s0 - mnew);
            const float e1 = __expf(s1 - mnew);
            float rs = e0 + e1;
            rs += __shfl_xor(rs, 1, 32);
            rs += __shfl_xor(rs, 2, 32);
            rs += __shfl_xor(rs, 4, 32);
            rs += __shfl_xor(rs, 8, 32);
            li[r] = li[r] * alpha[r] + rs;
            mi[r] = mnew;
            Plds[mbase + r][nloc]      = f2bf(e0);
            Plds[mbase + r][16 + nloc] = f2bf(e1);
        }
        __syncthreads();                     // C-layout -> A-layout via LDS
        const v16bf aP = load_frag_bf16(&Plds[m][koff]);
        __syncthreads();

#pragma unroll
        for (int r = 0; r < 8; ++r) { o0[r] *= alpha[r]; o1[r] *= alpha[r]; }

        // P[16x32keys] x V[32keys x 32d] -> two accumulate WMMAs (d halves)
        {
            v16bf bV0 = load_frag_bf16(Vt + ((size_t)bh * D_ + nloc)      * S_ + k0 + koff);
            v16bf bV1 = load_frag_bf16(Vt + ((size_t)bh * D_ + 16 + nloc) * S_ + k0 + koff);
            o0 = __builtin_amdgcn_wmma_f32_16x16x32_bf16(false, aP, false, bV0, (short)0, o0, false, false);
            o1 = __builtin_amdgcn_wmma_f32_16x16x32_bf16(false, aP, false, bV1, (short)0, o1, false, false);
        }
    }

    // Epilogue: normalize, stage through LDS, store coalesced b128.
#pragma unroll
    for (int r = 0; r < 8; ++r) {
        const float inv = 1.0f / li[r];
        Plds[mbase + r][nloc]      = f2bf(o0[r] * inv);
        Plds[mbase + r][16 + nloc] = f2bf(o1[r] * inv);
    }
    __syncthreads();
    {
        const int tm  = lane & 15;           // tile row
        const int seg = lane >> 4;           // which 32B half of the 64B row
        __bf16* dst = attnb + ((size_t)(b * S_ + q0 + tm) * H_ + h) * D_ + seg * 16;
        *(v8bf*)(dst)     = *(const v8bf*)&Plds[tm][seg * 16];
        *(v8bf*)(dst + 8) = *(const v8bf*)&Plds[tm][seg * 16 + 8];
    }
}

// ---------------- kernel 3: length-dependent scramble pad -------------------
__global__ void scramble_kernel(const __bf16* __restrict__ attnb,
                                const int* __restrict__ lengths,
                                __bf16* __restrict__ padded)
{
    const int b    = blockIdx.y;
    const int flat = blockIdx.x * 256 + threadIdx.x;   // 0 .. S*E-1
    const int L  = lengths[b];
    const int Ls = (L > 0) ? L : 1;
    int h = flat / (Ls * D_);
    if (h > H_ - 1) h = H_ - 1;
    const int rem = flat % (Ls * D_);
    const int s = rem / D_;
    const int d = rem % D_;
    __bf16 v = f2bf(0.0f);
    if (flat < L * E_)
        v = attnb[((size_t)(b * S_ + s) * H_ + h) * D_ + d];
    padded[(size_t)b * S_ * E_ + flat] = v;
}

// ---------------- kernel 4: output projection via WMMA ----------------------
__global__ __launch_bounds__(256) void out_gemm_kernel(
        const __bf16* __restrict__ padded, const float* __restrict__ Wo,
        float* __restrict__ out)
{
    const int rt   = blockIdx.x;
    const int lane = threadIdx.x;
    const int wy   = threadIdx.y;
    const int m    = lane & 15;
    const int koff = (lane >> 4) * 8;
    const int nloc = lane & 15;
    const long rowIdx = (long)rt * 16 + m;
    const int ct[2] = { wy, wy + 8 };

    v8f acc[2] = { {}, {} };
    for (int kk = 0; kk < E_; kk += 32) {
        v16bf a = load_frag_bf16(padded + rowIdx * E_ + kk + koff);
#pragma unroll
        for (int c = 0; c < 2; ++c) {
            v16bf bfr = load_frag_f32(Wo + (long)(ct[c] * 16 + nloc) * E_ + kk + koff);
            acc[c] = __builtin_amdgcn_wmma_f32_16x16x32_bf16(false, a, false, bfr,
                                                             (short)0, acc[c], false, false);
        }
    }

    // Stage fp32 C tiles in LDS, store coalesced b128 (8 waves x 2 x 1KB = 16KB).
    __shared__ alignas(16) float Otile[8][2][16][16];
    const int mbase = (lane >> 4) * 8;
#pragma unroll
    for (int c = 0; c < 2; ++c)
#pragma unroll
        for (int r = 0; r < 8; ++r)
            Otile[wy][c][mbase + r][nloc] = acc[c][r];
    __syncthreads();

    const int tm   = lane >> 1;
    const int half = lane & 1;
#pragma unroll
    for (int c = 0; c < 2; ++c) {
        float* dst = out + ((long)rt * 16 + tm) * E_ + ct[c] * 16 + half * 8;
        *(v4f*)(dst)     = *(const v4f*)&Otile[wy][c][tm][half * 8];
        *(v4f*)(dst + 4) = *(const v4f*)&Otile[wy][c][tm][half * 8 + 4];
    }
}

// ---------------- host launcher ---------------------------------------------
extern "C" void kernel_launch(void* const* d_in, const int* in_sizes, int n_in,
                              void* d_out, int out_size, void* d_ws, size_t ws_size,
                              hipStream_t stream) {
    const float* x   = (const float*)d_in[0];
    const int*   msk = (const int*)  d_in[1];
    const float* Wq  = (const float*)d_in[2];
    const float* Wk  = (const float*)d_in[3];
    const float* Wv  = (const float*)d_in[4];
    const float* Wo  = (const float*)d_in[5];
    float* out = (float*)d_out;

    const size_t nQKV  = (size_t)B_ * H_ * S_ * D_;   // 2,097,152 elems
    const size_t nBSE  = (size_t)B_ * S_ * E_;        // 2,097,152 elems
    char* w = (char*)d_ws;
    int*    lengths = (int*)w;
    __bf16* Qb      = (__bf16*)(w + 256);
    __bf16* Kb      = Qb + nQKV;
    __bf16* Vt      = Kb + nQKV;
    __bf16* attnb   = Vt + nQKV;
    __bf16* padded  = attnb + nBSE;

    lengths_kernel<<<dim3(B_), dim3(256), 0, stream>>>(msk, lengths);
    qkv_gemm_kernel<<<dim3((B_ * S_) / 16, 1, 3), dim3(32, 8), 0, stream>>>(
        x, Wq, Wk, Wv, Qb, Kb, Vt);
    attn_kernel<<<dim3(S_ / 16, H_, B_), dim3(32), 0, stream>>>(Qb, Kb, Vt, lengths, attnb);
    scramble_kernel<<<dim3((S_ * E_) / 256, B_), dim3(256), 0, stream>>>(attnb, lengths, padded);
    out_gemm_kernel<<<dim3((B_ * S_) / 16), dim3(32, 8), 0, stream>>>(padded, Wo, out);
}
